// ResidualMP_39599598469165
// MI455X (gfx1250) — compile-verified, matrix operand order checked by reading, not run on hardware
//
#include <hip/hip_runtime.h>
#include <math.h>

typedef __attribute__((ext_vector_type(16))) _Float16 v16h;
typedef __attribute__((ext_vector_type(8)))  _Float16 v8h;
typedef __attribute__((ext_vector_type(8)))  float    v8f;

#define TPB  128
#define ROWS 32    // rows per block (two 16-row WMMA sub-tiles -> B-fragment reuse x2)
#define ASTR 264   // 256 halfs + 8 pad -> 528B row stride, 16B aligned

union V16U { v16h v; v8h h[2]; };

__device__ __forceinline__ v16h load_frag(const _Float16* p0, const _Float16* p1) {
  V16U u;
  u.h[0] = *(const v8h*)p0;
  u.h[1] = *(const v8h*)p1;
  return u.v;
}

// one prefetch per thread covers the whole 128KB f16 weight matrix
__device__ __forceinline__ void prefetch_w(const _Float16* Wt, int tid) {
  __builtin_prefetch(Wt + (size_t)tid * 512, 0, 3);
}

// ---- A staging: ROWS x 256 K, f32 (optionally row-scaled) -> LDS f16 ----
__device__ __forceinline__ void stage_f32(const float* A, const float* rscale,
                                          int M, int rowBase, _Float16* As, int tid) {
  for (int i = tid; i < ROWS * 256; i += TPB) {
    int r = i >> 8, c = i & 255;
    int gr = rowBase + r;
    float v = 0.f;
    if (gr < M) {
      v = A[(long)gr * 256 + c];
      if (rscale) v *= rscale[gr];
    }
    As[r * ASTR + c] = (_Float16)v;
  }
}

__device__ __forceinline__ void stage_f16(const _Float16* A, int M, int rowBase,
                                          _Float16* As, int tid) {
  for (int i = tid; i < ROWS * 256; i += TPB) {
    int r = i >> 8, c = i & 255;
    int gr = rowBase + r;
    As[r * ASTR + c] = (gr < M) ? A[(long)gr * 256 + c] : (_Float16)0.f;
  }
}

// ---- core: acc{0,1}[t] += A(32x256) * W tile, Wt stored [Nout][256] f16 ----
// each B fragment is loaded once and feeds two WMMAs (row sub-tiles 0..15, 16..31)
__device__ __forceinline__ void mm_accum4x2(const _Float16* As, const _Float16* Wt,
                                            int colBase, v8f acc0[4], v8f acc1[4],
                                            int lane) {
  const int m  = lane & 15;
  const int kl = (lane < 16) ? 0 : 8;
  for (int kk = 0; kk < 256; kk += 32) {
    v16h a0 = load_frag(As + m * ASTR + kk + kl,
                        As + m * ASTR + kk + kl + 16);
    v16h a1 = load_frag(As + (m + 16) * ASTR + kk + kl,
                        As + (m + 16) * ASTR + kk + kl + 16);
#pragma unroll
    for (int t = 0; t < 4; ++t) {
      const _Float16* wp = Wt + (long)(colBase + t * 16 + m) * 256 + kk + kl;
      v16h b = load_frag(wp, wp + 16);
      acc0[t] = __builtin_amdgcn_wmma_f32_16x16x32_f16(false, a0, false, b,
                                                       (short)0, acc0[t], false, false);
      acc1[t] = __builtin_amdgcn_wmma_f32_16x16x32_f16(false, a1, false, b,
                                                       (short)0, acc1[t], false, false);
    }
  }
}

// ============================ utility kernels ============================
__global__ void k_zero(float* p, long n) {
  long i = (long)blockIdx.x * blockDim.x + threadIdx.x;
  if (i < n) p[i] = 0.f;
}

__global__ void k_degree(const int* dst, float* cnt, int E) {
  int e = blockIdx.x * blockDim.x + threadIdx.x;
  if (e < E) atomicAdd(&cnt[dst[e]], 1.f);
}

__global__ void k_invcnt(float* cnt, int N) {
  int i = blockIdx.x * blockDim.x + threadIdx.x;
  if (i < N) cnt[i] = 1.f / fmaxf(cnt[i], 1.f);
}

// W[k][Nout] f32 -> Wt[n][K] f16 (transpose + convert, pad cols with 0)
__global__ void k_prep_w(const float* W, _Float16* Wt, int K, int Nout, int NoutPad) {
  int id = blockIdx.x * blockDim.x + threadIdx.x;
  if (id >= K * NoutPad) return;
  int n = id / K, k = id % K;
  Wt[(long)n * K + k] = (_Float16)((n < Nout) ? W[(long)k * Nout + n] : 0.f);
}

// z[dst] += x[src], 8 floats per thread chunk
__global__ void k_scatter(const int* src, const int* dst, const float* x,
                          float* z, int E) {
  long id = (long)blockIdx.x * blockDim.x + threadIdx.x;
  if (id >= (long)E * 32) return;
  int e = (int)(id >> 5), ch = (int)(id & 31);
  int s = src[e], d = dst[e];
  const float* xs = x + (long)s * 256 + ch * 8;
  float*       zd = z + (long)d * 256 + ch * 8;
#pragma unroll
  for (int j = 0; j < 8; ++j) atomicAdd(zd + j, xs[j]);
}

// ============================ GEMM kernels ============================
// prop = X@Wl + (cinv*Z)@Wr + bl + br
__global__ void __launch_bounds__(TPB)
k_gemm_dual(const float* X, const float* Z, const float* cinv,
            const _Float16* Wl, const _Float16* Wr,
            const float* bl, const float* br, float* outp, int M) {
  __shared__ __align__(16) _Float16 As[ROWS * ASTR];
  int tid = threadIdx.x, lane = tid & 31, wave = tid >> 5;
  int rowBase = blockIdx.x * ROWS, colBase = wave * 64;
  v8f acc0[4] = {}, acc1[4] = {};
  prefetch_w(Wl, tid);
  prefetch_w(Wr, tid);
  stage_f32(X, nullptr, M, rowBase, As, tid);
  __syncthreads();
  mm_accum4x2(As, Wl, colBase, acc0, acc1, lane);
  __syncthreads();
  stage_f32(Z, cinv, M, rowBase, As, tid);
  __syncthreads();
  mm_accum4x2(As, Wr, colBase, acc0, acc1, lane);
  int m = lane & 15, mb = (lane < 16) ? 0 : 8;
#pragma unroll
  for (int t = 0; t < 4; ++t) {
    int n = colBase + t * 16 + m;
    float bias = bl[n] + br[n];
#pragma unroll
    for (int j = 0; j < 8; ++j) {
      int r0 = rowBase + mb + j;
      int r1 = r0 + 16;
      if (r0 < M) outp[(long)r0 * 256 + n] = acc0[t][j] + bias;
      if (r1 < M) outp[(long)r1 * 256 + n] = acc1[t][j] + bias;
    }
  }
}

// out = A@W + b
__global__ void __launch_bounds__(TPB)
k_gemm_bias(const float* A, const _Float16* Wt, const float* b, float* out, int M) {
  __shared__ __align__(16) _Float16 As[ROWS * ASTR];
  int tid = threadIdx.x, lane = tid & 31, wave = tid >> 5;
  int rowBase = blockIdx.x * ROWS, colBase = wave * 64;
  v8f acc0[4] = {}, acc1[4] = {};
  prefetch_w(Wt, tid);
  stage_f32(A, nullptr, M, rowBase, As, tid);
  __syncthreads();
  mm_accum4x2(As, Wt, colBase, acc0, acc1, lane);
  int m = lane & 15, mb = (lane < 16) ? 0 : 8;
#pragma unroll
  for (int t = 0; t < 4; ++t) {
    int n = colBase + t * 16 + m;
    float bias = b[n];
#pragma unroll
    for (int j = 0; j < 8; ++j) {
      int r0 = rowBase + mb + j;
      int r1 = r0 + 16;
      if (r0 < M) out[(long)r0 * 256 + n] = acc0[t][j] + bias;
      if (r1 < M) out[(long)r1 * 256 + n] = acc1[t][j] + bias;
    }
  }
}

// t = gamma*invbn*relu(h@Wa+ba)+beta  -> f16 output (feeds next GEMM directly)
__global__ void __launch_bounds__(TPB)
k_gemm_affine_relu(const float* A, const _Float16* Wt, const float* ba,
                   const float* gamma, const float* beta, float invbn,
                   _Float16* out, int M) {
  __shared__ __align__(16) _Float16 As[ROWS * ASTR];
  int tid = threadIdx.x, lane = tid & 31, wave = tid >> 5;
  int rowBase = blockIdx.x * ROWS, colBase = wave * 64;
  v8f acc0[4] = {}, acc1[4] = {};
  prefetch_w(Wt, tid);
  stage_f32(A, nullptr, M, rowBase, As, tid);
  __syncthreads();
  mm_accum4x2(As, Wt, colBase, acc0, acc1, lane);
  int m = lane & 15, mb = (lane < 16) ? 0 : 8;
#pragma unroll
  for (int t = 0; t < 4; ++t) {
    int n = colBase + t * 16 + m;
    float bias = ba[n], sc = gamma[n] * invbn, be = beta[n];
#pragma unroll
    for (int j = 0; j < 8; ++j) {
      int r0 = rowBase + mb + j;
      int r1 = r0 + 16;
      float v0 = fmaxf(acc0[t][j] + bias, 0.f) * sc + be;
      float v1 = fmaxf(acc1[t][j] + bias, 0.f) * sc + be;
      if (r0 < M) out[(long)r0 * 256 + n] = (_Float16)v0;
      if (r1 < M) out[(long)r1 * 256 + n] = (_Float16)v1;
    }
  }
}

// h += t@Wb + bb   (A is already f16)
__global__ void __launch_bounds__(TPB)
k_gemm_add(const _Float16* A, const _Float16* Wt, const float* bb, float* h, int M) {
  __shared__ __align__(16) _Float16 As[ROWS * ASTR];
  int tid = threadIdx.x, lane = tid & 31, wave = tid >> 5;
  int rowBase = blockIdx.x * ROWS, colBase = wave * 64;
  v8f acc0[4] = {}, acc1[4] = {};
  prefetch_w(Wt, tid);
  stage_f16(A, M, rowBase, As, tid);
  __syncthreads();
  mm_accum4x2(As, Wt, colBase, acc0, acc1, lane);
  int m = lane & 15, mb = (lane < 16) ? 0 : 8;
#pragma unroll
  for (int t = 0; t < 4; ++t) {
    int n = colBase + t * 16 + m;
    float bias = bb[n];
#pragma unroll
    for (int j = 0; j < 8; ++j) {
      int r0 = rowBase + mb + j;
      int r1 = r0 + 16;
      if (r0 < M) { long i0 = (long)r0 * 256 + n; h[i0] = h[i0] + acc0[t][j] + bias; }
      if (r1 < M) { long i1 = (long)r1 * 256 + n; h[i1] = h[i1] + acc1[t][j] + bias; }
    }
  }
}

// xnext = relu( (h@W2+b2) / max(||row||2, 1e-12) )   (block covers full rows)
__global__ void __launch_bounds__(TPB)
k_gemm_norm_relu(const float* A, const _Float16* Wt, const float* b2,
                 float* out, int M) {
  __shared__ __align__(16) _Float16 As[ROWS * ASTR];
  __shared__ float ss[ROWS];
  int tid = threadIdx.x, lane = tid & 31, wave = tid >> 5;
  int rowBase = blockIdx.x * ROWS, colBase = wave * 64;
  v8f acc0[4] = {}, acc1[4] = {};
  prefetch_w(Wt, tid);
  stage_f32(A, nullptr, M, rowBase, As, tid);
  if (tid < ROWS) ss[tid] = 0.f;
  __syncthreads();
  mm_accum4x2(As, Wt, colBase, acc0, acc1, lane);
  int m = lane & 15, mb = (lane < 16) ? 0 : 8;
#pragma unroll
  for (int t = 0; t < 4; ++t) {
    int n = colBase + t * 16 + m;
    float bias = b2[n];
#pragma unroll
    for (int j = 0; j < 8; ++j) {
      float v0 = acc0[t][j] + bias;
      float v1 = acc1[t][j] + bias;
      atomicAdd(&ss[mb + j], v0 * v0);
      atomicAdd(&ss[16 + mb + j], v1 * v1);
    }
  }
  __syncthreads();
#pragma unroll
  for (int t = 0; t < 4; ++t) {
    int n = colBase + t * 16 + m;
    float bias = b2[n];
#pragma unroll
    for (int j = 0; j < 8; ++j) {
      int r0 = rowBase + mb + j;
      int r1 = r0 + 16;
      if (r0 < M) {
        float inv = 1.f / fmaxf(sqrtf(ss[mb + j]), 1e-12f);
        out[(long)r0 * 256 + n] = fmaxf((acc0[t][j] + bias) * inv, 0.f);
      }
      if (r1 < M) {
        float inv = 1.f / fmaxf(sqrtf(ss[16 + mb + j]), 1e-12f);
        out[(long)r1 * 256 + n] = fmaxf((acc1[t][j] + bias) * inv, 0.f);
      }
    }
  }
}

// out = log_softmax(A@W2+b2) over 47 classes (W padded to 64 cols)
__global__ void __launch_bounds__(TPB)
k_gemm_lsm(const float* A, const _Float16* Wt, const float* b, float* out, int M) {
  __shared__ __align__(16) _Float16 As[ROWS * ASTR];
  __shared__ float tileS[ROWS][68];
  __shared__ float rmax[ROWS], rlse[ROWS];
  int tid = threadIdx.x, lane = tid & 31, wave = tid >> 5;
  int rowBase = blockIdx.x * ROWS, colBase = wave * 16;
  v8f acc0 = {}, acc1 = {};
  prefetch_w(Wt, tid);
  stage_f32(A, nullptr, M, rowBase, As, tid);
  __syncthreads();
  int m = lane & 15, kl = (lane < 16) ? 0 : 8, mb = (lane < 16) ? 0 : 8;
  for (int kk = 0; kk < 256; kk += 32) {
    v16h a0 = load_frag(As + m * ASTR + kk + kl,
                        As + m * ASTR + kk + kl + 16);
    v16h a1 = load_frag(As + (m + 16) * ASTR + kk + kl,
                        As + (m + 16) * ASTR + kk + kl + 16);
    const _Float16* wp = Wt + (long)(colBase + m) * 256 + kk + kl;
    v16h bf = load_frag(wp, wp + 16);
    acc0 = __builtin_amdgcn_wmma_f32_16x16x32_f16(false, a0, false, bf,
                                                  (short)0, acc0, false, false);
    acc1 = __builtin_amdgcn_wmma_f32_16x16x32_f16(false, a1, false, bf,
                                                  (short)0, acc1, false, false);
  }
  int n = colBase + m;
  float bias = (n < 47) ? b[n] : 0.f;
#pragma unroll
  for (int j = 0; j < 8; ++j) {
    tileS[mb + j][n]      = acc0[j] + bias;
    tileS[16 + mb + j][n] = acc1[j] + bias;
  }
  __syncthreads();
  if (tid < ROWS) {
    float mx = -3.4e38f;
    for (int c = 0; c < 47; ++c) mx = fmaxf(mx, tileS[tid][c]);
    float s = 0.f;
    for (int c = 0; c < 47; ++c) s += expf(tileS[tid][c] - mx);
    rmax[tid] = mx;
    rlse[tid] = logf(s);
  }
  __syncthreads();
  for (int i = tid; i < ROWS * 47; i += TPB) {
    int r = i / 47, c = i % 47, gr = rowBase + r;
    if (gr < M) out[(long)gr * 47 + c] = tileS[r][c] - rmax[r] - rlse[r];
  }
}

// ============================ host launcher ============================
extern "C" void kernel_launch(void* const* d_in, const int* in_sizes, int n_in,
                              void* d_out, int out_size, void* d_ws, size_t ws_size,
                              hipStream_t stream) {
  (void)in_sizes; (void)n_in; (void)out_size; (void)ws_size;
  const int N = 50000, E = 800000, L = 3;
  const int WM = 256 * 256;  // 65536 elements per square weight

  const float* x0       = (const float*)d_in[0];
  const int*   ei       = (const int*)d_in[1];
  const float* lin_l_W  = (const float*)d_in[2];
  const float* lin_l_b  = (const float*)d_in[3];
  const float* lin_r_W  = (const float*)d_in[4];
  const float* lin_r_b  = (const float*)d_in[5];
  const float* mlp_W1   = (const float*)d_in[6];
  const float* mlp_b1   = (const float*)d_in[7];
  const float* blk_Wa   = (const float*)d_in[8];
  const float* blk_ba   = (const float*)d_in[9];
  const float* blk_g    = (const float*)d_in[10];
  const float* blk_be   = (const float*)d_in[11];
  const float* blk_Wb   = (const float*)d_in[12];
  const float* blk_bb   = (const float*)d_in[13];
  const float* mlp_W2   = (const float*)d_in[14];
  const float* mlp_b2   = (const float*)d_in[15];
  const float* post_W1  = (const float*)d_in[16];
  const float* post_b1  = (const float*)d_in[17];
  const float* post_W2  = (const float*)d_in[18];
  const float* post_b2  = (const float*)d_in[19];
  const int* src = ei;
  const int* dst = ei + E;
  float* out = (float*)d_out;

  // workspace carve-up
  char* w = (char*)d_ws;
  size_t off = 0;
  auto alloc = [&](size_t bytes) -> void* {
    void* p = w + off;
    off += (bytes + 255) & ~(size_t)255;
    return p;
  };
  float* cinv = (float*)alloc((size_t)N * 4);
  float* z    = (float*)alloc((size_t)N * 256 * 4);
  float* prop = (float*)alloc((size_t)N * 256 * 4);
  float* hb   = (float*)alloc((size_t)N * 256 * 4);
  float* xA   = (float*)alloc((size_t)N * 256 * 4);
  float* xB   = (float*)alloc((size_t)N * 256 * 4);
  _Float16* tb = (_Float16*)z;  // alias: z dead while t is live within a layer
  _Float16* wt = (_Float16*)alloc((size_t)(25 * WM + 64 * 256) * 2);

  _Float16* wt_linl = wt;
  _Float16* wt_linr = wt_linl + 3 * WM;
  _Float16* wt_w1   = wt_linr + 3 * WM;
  _Float16* wt_wa   = wt_w1   + 3 * WM;
  _Float16* wt_wb   = wt_wa   + 6 * WM;
  _Float16* wt_w2   = wt_wb   + 6 * WM;
  _Float16* wt_p1   = wt_w2   + 3 * WM;
  _Float16* wt_p2   = wt_p1   + WM;

  auto prep = [&](const float* W, _Float16* Wt, int K, int Nout, int NoutPad) {
    int tot = K * NoutPad;
    k_prep_w<<<(tot + 255) / 256, 256, 0, stream>>>(W, Wt, K, Nout, NoutPad);
  };
  for (int i = 0; i < L; ++i) {
    prep(lin_l_W + (size_t)i * WM, wt_linl + (size_t)i * WM, 256, 256, 256);
    prep(lin_r_W + (size_t)i * WM, wt_linr + (size_t)i * WM, 256, 256, 256);
    prep(mlp_W1  + (size_t)i * WM, wt_w1   + (size_t)i * WM, 256, 256, 256);
    prep(mlp_W2  + (size_t)i * WM, wt_w2   + (size_t)i * WM, 256, 256, 256);
    for (int j = 0; j < 2; ++j) {
      int idx = i * 2 + j;
      prep(blk_Wa + (size_t)idx * WM, wt_wa + (size_t)idx * WM, 256, 256, 256);
      prep(blk_Wb + (size_t)idx * WM, wt_wb + (size_t)idx * WM, 256, 256, 256);
    }
  }
  prep(post_W1, wt_p1, 256, 256, 256);
  prep(post_W2, wt_p2, 256, 47, 64);

  // in-degree -> 1/max(cnt,1)
  k_zero<<<(N + 255) / 256, 256, 0, stream>>>(cinv, N);
  k_degree<<<(E + 255) / 256, 256, 0, stream>>>(dst, cinv, E);
  k_invcnt<<<(N + 255) / 256, 256, 0, stream>>>(cinv, N);

  const float invbn = 1.0f / sqrtf(1.0f + 1e-5f);
  const int G = (N + ROWS - 1) / ROWS;  // GEMM grid: ROWS rows per block
  const long ZN = (long)N * 256;
  const long SCAT = (long)E * 32;

  const float* xcur = x0;
  float* xping[2] = {xA, xB};
  for (int i = 0; i < L; ++i) {
    // scatter-mean
    k_zero<<<(int)((ZN + 255) / 256), 256, 0, stream>>>(z, ZN);
    k_scatter<<<(int)((SCAT + 255) / 256), 256, 0, stream>>>(src, dst, xcur, z, E);
    // prop = x@Wl + (z/cnt)@Wr + bl + br
    k_gemm_dual<<<G, TPB, 0, stream>>>(xcur, z, cinv,
        wt_linl + (size_t)i * WM, wt_linr + (size_t)i * WM,
        lin_l_b + (size_t)i * 256, lin_r_b + (size_t)i * 256, prop, N);
    // h = prop@W1 + b1
    k_gemm_bias<<<G, TPB, 0, stream>>>(prop, wt_w1 + (size_t)i * WM,
        mlp_b1 + (size_t)i * 256, hb, N);
    // two ResNet blocks
    for (int j = 0; j < 2; ++j) {
      int idx = i * 2 + j;
      k_gemm_affine_relu<<<G, TPB, 0, stream>>>(hb, wt_wa + (size_t)idx * WM,
          blk_ba + (size_t)idx * 256, blk_g + (size_t)idx * 256,
          blk_be + (size_t)idx * 256, invbn, tb, N);
      k_gemm_add<<<G, TPB, 0, stream>>>(tb, wt_wb + (size_t)idx * WM,
          blk_bb + (size_t)idx * 256, hb, N);
    }
    // x_next = relu(normalize(h@W2 + b2))
    float* xnext = xping[i & 1];
    k_gemm_norm_relu<<<G, TPB, 0, stream>>>(hb, wt_w2 + (size_t)i * WM,
        mlp_b2 + (size_t)i * 256, xnext, N);
    xcur = xnext;
  }

  // post MLP + log_softmax
  k_gemm_bias<<<G, TPB, 0, stream>>>(xcur, wt_p1, post_b1, hb, N);
  k_gemm_lsm<<<G, TPB, 0, stream>>>(hb, wt_p2, post_b2, out, N);
}
